// AttentionConv_8658654069070
// MI455X (gfx1250) — compile-verified, matrix operand order checked by reading, not run on hardware
//
#include <hip/hip_runtime.h>
#include <hip/hip_bf16.h>
#include <math.h>

// ---- problem constants (from reference) ----
#define B_   4
#define C_   128
#define N_   2048
#define K_   16
#define G_   4
#define CLG  24      // L_CH / G
#define CNG  8       // NL_CH / G
#define MT_  18      // 288 / 16 stacked [Wq;Wk;Wv] row tiles
#define KS_  4       // 128 / 32 k-steps

typedef __bf16 v16bf __attribute__((ext_vector_type(16)));
typedef float  v8f   __attribute__((ext_vector_type(8)));

// (lane, elem j) -> K index inside a 32-wide K step, used identically for the
// A and B bf16 fragments (CDNA5 ISA 16-bit A layout: lanes 0-15 hold
// K {0..7,16..23}, lanes 16-31 hold K {8..15,24..31}, pairs per VGPR).
__device__ __forceinline__ int kmap(int lane, int j) {
  return (j < 8 ? j : j + 8) + ((lane & 16) ? 8 : 0);
}

// ---------------------------------------------------------------------------
// Kernel 0: zero score buffer (must happen every call — atomics accumulate),
// and build bf16 A-fragments for the stacked [Wq;Wk;Wv] (288x128) weights.
// Afrag layout: [mtile][kstep][lane][16 elems], 32B per lane -> one v16bf load.
// ---------------------------------------------------------------------------
__global__ void prep_kernel(const float* __restrict__ Wq,
                            const float* __restrict__ Wk,
                            const float* __restrict__ Wv,
                            float* __restrict__ score,
                            __bf16* __restrict__ Afrag) {
  int e = blockIdx.x * blockDim.x + threadIdx.x;   // 0 .. 36863
  if (e < B_ * G_ * N_) score[e] = 0.0f;           // 32768 floats
  if (e < MT_ * KS_ * 32 * 16) {
    int j    = e & 15;
    int lane = (e >> 4) & 31;
    int ks   = (e >> 9) & 3;
    int mt   = e >> 11;
    int m = mt * 16 + (lane & 15);
    int c = ks * 32 + kmap(lane, j);
    float w;
    if      (m < 96)  w = Wq[m * C_ + c];
    else if (m < 192) w = Wk[(m - 96) * C_ + c];
    else              w = Wv[(m - 192) * C_ + c];
    Afrag[e] = (__bf16)w;
  }
}

// ---------------------------------------------------------------------------
// Kernel 1: fused QKV GEMM (bf16 WMMA, f32 acc) + softmax + out_l + score
// scatter. One wave per n: 18 M-tiles x 4 K-steps = 72 v_wmma_f32_16x16x32_bf16.
// Block = 64 threads (2 waves).  LDS: 2 x 288x17 f32 staging (~39 KB).
// ---------------------------------------------------------------------------
__global__ void __launch_bounds__(64)
qkv_attn_kernel(const float* __restrict__ x,
                const int*   __restrict__ idx,
                const __bf16* __restrict__ Afrag,
                float* __restrict__ score,
                float* __restrict__ dout) {
  __shared__ float qkv[2][288 * 17];   // [wave][m*17 + k] padded vs bank conflicts
  __shared__ float obuf[2][16];

  const int lane = threadIdx.x & 31;
  const int w    = threadIdx.x >> 5;
  const int wid  = blockIdx.x * 2 + w;       // 0 .. B*N-1
  const int b    = wid >> 11;                // / N_
  const int n    = wid & (N_ - 1);
  const int kcol = lane & 15;

  // L2-resident weight fragments: hint the prefetcher (global_prefetch_b8)
  __builtin_prefetch(Afrag, 0, 3);

  // --- build B fragments: x[b, c, n, k] columns for this n, f32 -> bf16 ---
  v16bf bfrag[KS_];
  #pragma unroll
  for (int ks = 0; ks < KS_; ++ks) {
    #pragma unroll
    for (int j = 0; j < 16; ++j) {
      int c = ks * 32 + kmap(lane, j);
      bfrag[ks][j] = (__bf16)x[(((b * C_ + c) * N_) + n) * K_ + kcol];
    }
  }

  // --- GEMM: 18 output tiles of 16 channels x 16 k ---
  for (int mt = 0; mt < MT_; ++mt) {
    v8f acc = {};
    #pragma unroll
    for (int ks = 0; ks < KS_; ++ks) {
      v16bf af = *(const v16bf*)(Afrag + ((mt * KS_ + ks) * 32 + lane) * 16);
      acc = __builtin_amdgcn_wmma_f32_16x16x32_bf16(
          false, af, false, bfrag[ks], (short)0, acc, false, false);
    }
    // C/D layout: lane<16 -> rows r, lane>=16 -> rows r+8; col = lane&15
    int mbase = mt * 16 + ((lane >> 4) << 3);
    #pragma unroll
    for (int r = 0; r < 8; ++r)
      qkv[w][(mbase + r) * 17 + kcol] = acc[r];
  }
  __syncthreads();

  // --- epilogue per group: softmax over K=16, scatter score, out_l ---
  for (int g = 0; g < G_; ++g) {
    float s = 0.0f;
    #pragma unroll
    for (int c = 0; c < CLG; ++c)
      s += qkv[w][(g * CLG + c) * 17 + kcol] *
           qkv[w][(96 + g * CLG + c) * 17 + kcol];
    // lanes 16-31 mirror lanes 0-15 (same kcol) -> full-wave butterflies OK
    float mx = s;
    #pragma unroll
    for (int m = 1; m < 16; m <<= 1) mx = fmaxf(mx, __shfl_xor(mx, m, 32));
    float e = __expf(s - mx);
    float sum = e;
    #pragma unroll
    for (int m = 1; m < 16; m <<= 1) sum += __shfl_xor(sum, m, 32);
    float o = e / sum;

    if (lane < 16) {
      atomicAdd(&score[(b * G_ + g) * N_ + idx[(b * N_ + n) * K_ + kcol]], o);
      obuf[w][kcol] = o;
    }
    __syncthreads();
    if (lane < CLG) {
      float a2 = 0.0f;
      #pragma unroll
      for (int k2 = 0; k2 < 16; ++k2)
        a2 += obuf[w][k2] * qkv[w][(192 + g * CLG + lane) * 17 + k2];
      dout[((b * 128) + g * CLG + lane) * N_ + n] = a2;   // channels 0..95
    }
    __syncthreads();
  }
}

// ---------------------------------------------------------------------------
// Kernel 2: per (b,g) top-16 of score (lowest-index tie-break like lax.top_k),
// then precompute kn_g and tanh-scaled vn_g (8x16 per (b,g)).
// ---------------------------------------------------------------------------
__global__ void __launch_bounds__(256)
topk_prep_kernel(const float* __restrict__ score,
                 const float* __restrict__ absx,
                 const float* __restrict__ Wnk,
                 const float* __restrict__ Wnv,
                 float* __restrict__ knbuf,
                 float* __restrict__ vnbuf) {
  __shared__ float sc[N_];
  __shared__ float rv[256];
  __shared__ int   ri[256];
  __shared__ int   sidx[16];
  __shared__ float stanh[16];

  const int tid = threadIdx.x;
  const int bg = blockIdx.x;                 // 0..15
  const int b = bg >> 2, g = bg & 3;

  for (int i = tid; i < N_; i += 256) sc[i] = score[bg * N_ + i];
  __syncthreads();

  for (int r = 0; r < K_; ++r) {
    float bv = -INFINITY; int bi = 1 << 30;
    for (int i = tid; i < N_; i += 256) {
      float v = sc[i];
      if (v > bv || (v == bv && i < bi)) { bv = v; bi = i; }
    }
    rv[tid] = bv; ri[tid] = bi;
    __syncthreads();
    for (int s2 = 128; s2 > 0; s2 >>= 1) {
      if (tid < s2) {
        float v2 = rv[tid + s2]; int i2 = ri[tid + s2];
        if (v2 > rv[tid] || (v2 == rv[tid] && i2 < ri[tid])) {
          rv[tid] = v2; ri[tid] = i2;
        }
      }
      __syncthreads();
    }
    if (tid == 0) {
      sidx[r]  = ri[0];
      stanh[r] = tanhf(rv[0]);
      sc[ri[0]] = -INFINITY;
    }
    __syncthreads();
  }

  if (tid < 128) {                            // (c,j): 8 x 16
    int c = tid >> 4, j = tid & 15;
    int nj = sidx[j];
    float th = stanh[j];
    float kk = 0.0f, vv = 0.0f;
    for (int cc = 0; cc < C_ / 2; ++cc) {
      float a = absx[(b * (C_ / 2) + cc) * N_ + nj];
      kk += Wnk[(g * CNG + c) * (C_ / 2) + cc] * a;
      vv += Wnv[(g * CNG + c) * (C_ / 2) + cc] * a;
    }
    knbuf[(bg * CNG + c) * 16 + j] = kk;
    vnbuf[(bg * CNG + c) * 16 + j] = vv * th;
  }
}

// ---------------------------------------------------------------------------
// Kernel 3: non-local attention, one thread per (b,g,n). ~0.05 GFLOP total.
// ---------------------------------------------------------------------------
__global__ void __launch_bounds__(256)
nonlocal_kernel(const float* __restrict__ absx,
                const float* __restrict__ Wnq,
                const float* __restrict__ knbuf,
                const float* __restrict__ vnbuf,
                float* __restrict__ dout) {
  int flat = blockIdx.x * 256 + threadIdx.x;   // 0 .. B*G*N-1
  int b = flat >> 13;                          // / (G*N)
  int g = (flat >> 11) & 3;
  int n = flat & (N_ - 1);
  int bg = b * G_ + g;

  float qn[CNG];
  #pragma unroll
  for (int c = 0; c < CNG; ++c) qn[c] = 0.0f;
  for (int cc = 0; cc < C_ / 2; ++cc) {
    float a = absx[(b * (C_ / 2) + cc) * N_ + n];
    #pragma unroll
    for (int c = 0; c < CNG; ++c)
      qn[c] += Wnq[(g * CNG + c) * (C_ / 2) + cc] * a;
  }

  float lg[16], mx = -INFINITY;
  #pragma unroll
  for (int j = 0; j < 16; ++j) {
    float s = 0.0f;
    #pragma unroll
    for (int c = 0; c < CNG; ++c) s += qn[c] * knbuf[(bg * CNG + c) * 16 + j];
    lg[j] = s;
    mx = fmaxf(mx, s);
  }
  float sum = 0.0f;
  #pragma unroll
  for (int j = 0; j < 16; ++j) { lg[j] = __expf(lg[j] - mx); sum += lg[j]; }
  float inv = 1.0f / sum;
  #pragma unroll
  for (int c = 0; c < CNG; ++c) {
    float o = 0.0f;
    #pragma unroll
    for (int j = 0; j < 16; ++j) o += lg[j] * vnbuf[(bg * CNG + c) * 16 + j];
    dout[(b * 128 + 96 + g * CNG + c) * N_ + n] = o * inv;   // channels 96..127
  }
}

// ---------------------------------------------------------------------------
extern "C" void kernel_launch(void* const* d_in, const int* in_sizes, int n_in,
                              void* d_out, int out_size, void* d_ws, size_t ws_size,
                              hipStream_t stream) {
  const float* x    = (const float*)d_in[0];
  const float* absx = (const float*)d_in[1];
  const int*   idx  = (const int*)  d_in[2];
  const float* Wq   = (const float*)d_in[3];
  const float* Wk   = (const float*)d_in[4];
  const float* Wv   = (const float*)d_in[5];
  const float* Wnq  = (const float*)d_in[6];
  const float* Wnk  = (const float*)d_in[7];
  const float* Wnv  = (const float*)d_in[8];
  float* dout = (float*)d_out;

  // workspace layout (all offsets 256B-aligned)
  char* ws = (char*)d_ws;
  float*  score = (float*) (ws + 0);           // B*G*N f32    = 131072 B
  __bf16* Afrag = (__bf16*)(ws + 131072);      // 18*4*32*16   =  73728 B
  float*  knbuf = (float*) (ws + 204800);      // B*G*8*16 f32 =   8192 B
  float*  vnbuf = (float*) (ws + 212992);      //               =   8192 B

  prep_kernel<<<144, 256, 0, stream>>>(Wq, Wk, Wv, score, Afrag);
  qkv_attn_kernel<<<(B_ * N_) / 2, 64, 0, stream>>>(x, idx, Afrag, score, dout);
  topk_prep_kernel<<<B_ * G_, 256, 0, stream>>>(score, absx, Wnk, Wnv, knbuf, vnbuf);
  nonlocal_kernel<<<(B_ * G_ * N_) / 256, 256, 0, stream>>>(absx, Wnq, knbuf, vnbuf, dout);
}
